// Aligner2_36885179138567
// MI455X (gfx1250) — matrix operand order repaired and checked by reading, not compiled, not run on hardware
//
#include <hip/hip_runtime.h>
#include <cstdint>

typedef __bf16 bf16;
typedef __attribute__((ext_vector_type(16))) __bf16 v16bf;
typedef __attribute__((ext_vector_type(8)))  __bf16 v8bf;
typedef __attribute__((ext_vector_type(8)))  float  v8f;

#define SLOPE       0.3f
#define TEMPERATURE 0.0005f

#if defined(__has_builtin)
#if __has_builtin(__builtin_amdgcn_sched_barrier)
#define SCHED_FENCE() __builtin_amdgcn_sched_barrier(0)
#endif
#endif
#ifndef SCHED_FENCE
#define SCHED_FENCE()
#endif

__device__ __forceinline__ float bf2f(bf16 x) {
  unsigned short u = __builtin_bit_cast(unsigned short, x);
  unsigned int v = ((unsigned int)u) << 16;
  return __builtin_bit_cast(float, v);
}
__device__ __forceinline__ bf16 f2bf(float f) {
  unsigned int u = __builtin_bit_cast(unsigned int, f);
  u += 0x7FFFu + ((u >> 16) & 1u);            // round-to-nearest-even
  unsigned short h = (unsigned short)(u >> 16);
  return __builtin_bit_cast(bf16, h);
}
__device__ __forceinline__ v8bf zero8() {
  v8bf z;
#pragma unroll
  for (int i = 0; i < 8; ++i) z[i] = __builtin_bit_cast(bf16, (unsigned short)0);
  return z;
}
__device__ __forceinline__ v16bf cat16(v8bf lo, v8bf hi) {
  return __builtin_shufflevector(lo, hi, 0,1,2,3,4,5,6,7,8,9,10,11,12,13,14,15);
}

// ---------------- tiled transpose + f32->bf16:  in [B][C][T] f32  ->  out [B][T][C] bf16 ----
__global__ void __launch_bounds__(256) transpose_f2bf(const float* __restrict__ in,
                                                      bf16* __restrict__ out,
                                                      int C, int T) {
  __shared__ float tile[32][33];
  const int b  = blockIdx.z;
  const int t0 = blockIdx.x * 32;
  const int c0 = blockIdx.y * 32;
  const float* inb = in + (size_t)b * C * T;
  bf16* outb = out + (size_t)b * T * C;
#pragma unroll
  for (int j = 0; j < 4; ++j) {
    int c = c0 + threadIdx.y + j * 8;
    int t = t0 + threadIdx.x;
    if (c < C && t < T) tile[threadIdx.y + j * 8][threadIdx.x] = inb[(size_t)c * T + t];
  }
  __syncthreads();
#pragma unroll
  for (int j = 0; j < 4; ++j) {
    int t = t0 + threadIdx.y + j * 8;
    int c = c0 + threadIdx.x;
    if (c < C && t < T) outb[(size_t)t * C + c] = f2bf(tile[threadIdx.x][threadIdx.y + j * 8]);
  }
}

// ---------------- weight permute + convert:  w[co][ci][kw] f32 -> wP[co][kw][ci] bf16 -------
__global__ void __launch_bounds__(256) permw_f2bf(const float* __restrict__ in,
                                                  bf16* __restrict__ out,
                                                  int Cin, int KW, int n) {
  int i = blockIdx.x * 256 + threadIdx.x;
  if (i >= n) return;
  const int ck = Cin * KW;
  const int co = i / ck, r = i - co * ck;
  const int kk = r / Cin, ci = r - kk * Cin;
  out[i] = f2bf(in[(size_t)co * ck + (size_t)ci * KW + kk]);
}

// ---------------- conv1d + bias + leaky-ReLU, implicit GEMM on WMMA, T-major I/O ------------
// XT: [B][TinRows][Cin] bf16 (rows t < Tin valid), WP: [256][KW*Cin] bf16 (kk-major),
// YT: [B][ToutRows][256] bf16. pad = 3. Each wave: 16(co) x 32(t), software-pipelined k-loop.
template<int KW>
__global__ void __launch_bounds__(128) conv_lrelu_wmma(
    const bf16* __restrict__ XT, const bf16* __restrict__ WP,
    const float* __restrict__ bias, bf16* __restrict__ YT,
    int Cin, int Tin, int TinRows, int ToutRows, int Ntiles)
{
  const int lane = threadIdx.x & 31;
  const int wv   = threadIdx.x >> 5;
  const int nt0  = (blockIdx.x * 4 + wv) * 2;
  if (nt0 >= Ntiles) return;                    // wave-uniform
  const bool has1 = (nt0 + 1) < Ntiles;         // wave-uniform
  const int mtile = blockIdx.y;
  const int b     = blockIdx.z;
  const int g  = lane >> 4;
  const int mn = lane & 15;
  const int Ktot   = Cin * KW;
  const int ksteps = (Ktot + 31) >> 5;
  const bf16* Arow = WP + (size_t)(mtile * 16 + mn) * Ktot;
  const bf16* Xb   = XT + (size_t)b * TinRows * Cin;
  const int t0 = nt0 * 16 + mn;
  const int t1 = t0 + 16;

  auto loadA = [&](int ks) -> v16bf {
    const int a0 = ks * 32 + g * 16, a1 = a0 + 8;   // Ktot multiple of 8 in every layer
    v8bf lo = (a0 < Ktot) ? *(const v8bf*)(Arow + a0) : zero8();
    v8bf hi = (a1 < Ktot) ? *(const v8bf*)(Arow + a1) : zero8();
    return cat16(lo, hi);
  };
  auto loadB = [&](int t, int kk, int ci0) -> v16bf {
    v8bf lo = zero8(), hi = zero8();
    if (kk < KW) {
      const int ts = t + kk - 3;                    // pad = 3
      if (ts >= 0 && ts < Tin) {
        const bf16* p = Xb + (size_t)ts * Cin + ci0;
        lo = *(const v8bf*)p; hi = *(const v8bf*)(p + 8);
      }
    }
    return cat16(lo, hi);
  };

  // (kk, ci0): this lane's B-run position; kidx0 = ks*32 + g*16 (kk-major composite index)
  int kk = 0, ci0 = g * 16;                         // g*16 < Cin for all layers
  v16bf av  = loadA(0);
  v16bf bv0 = loadB(t0, kk, ci0);
  v16bf bv1 = has1 ? loadB(t1, kk, ci0) : bv0;

  v8f acc0 = {}; v8f acc1 = {};
  for (int ks = 0; ks < ksteps; ++ks) {
    // advance k-position one step ahead and issue next-step loads BEFORE this step's WMMAs
    int kkn = kk, ci0n = ci0 + 32;
    if (ci0n >= Cin) { ci0n -= Cin; ++kkn; }        // 32 < Cin: never skips two blocks
    v16bf avN = av, bv0N = bv0, bv1N = bv1;
    if (ks + 1 < ksteps) {
      avN  = loadA(ks + 1);
      bv0N = loadB(t0, kkn, ci0n);
      if (has1) bv1N = loadB(t1, kkn, ci0n);
    }
    SCHED_FENCE();   // keep next-step loads above the WMMAs: they fly during matrix work
    acc0 = __builtin_amdgcn_wmma_f32_16x16x32_bf16(false, av, false, bv0,
                                                   (short)0, acc0, false, false);
    if (has1)
      acc1 = __builtin_amdgcn_wmma_f32_16x16x32_bf16(false, av, false, bv1,
                                                     (short)0, acc1, false, false);
    SCHED_FENCE();
    av = avN; bv0 = bv0N; bv1 = bv1N;
    kk = kkn; ci0 = ci0n;
  }

  // bias (rows co = mtile*16 + g*8 + r), leaky-ReLU, transposed b128 store
  float bvv[8];
#pragma unroll
  for (int r = 0; r < 8; ++r) bvv[r] = bias[mtile * 16 + g * 8 + r];
  bf16* Yb = YT + (size_t)b * ToutRows * 256 + mtile * 16 + g * 8;
  {
    v8bf o;
#pragma unroll
    for (int r = 0; r < 8; ++r) {
      float v = acc0[r] + bvv[r];
      o[r] = f2bf((v > 0.0f) ? v : SLOPE * v);
    }
    *(v8bf*)(Yb + (size_t)t0 * 256) = o;
  }
  if (has1) {
    v8bf o;
#pragma unroll
    for (int r = 0; r < 8; ++r) {
      float v = acc1[r] + bvv[r];
      o[r] = f2bf((v > 0.0f) ? v : SLOPE * v);
    }
    *(v8bf*)(Yb + (size_t)t1 * 256) = o;
  }
}

// ---------------- per-(b,t) sum of squares over 256 channels, T-major -----------------------
__global__ void __launch_bounds__(256) sumsqT_kernel(const bf16* __restrict__ X,
                                                     float* __restrict__ out,
                                                     int TrowsAlloc, int Tvalid, int total) {
  int i = blockIdx.x * 256 + threadIdx.x;
  if (i >= total) return;
  int b = i / Tvalid, t = i - b * Tvalid;
  const bf16* p = X + ((size_t)b * TrowsAlloc + t) * 256;
  float s = 0.0f;
  for (int c = 0; c < 256; c += 8) {
    v8bf v = *(const v8bf*)(p + c);
#pragma unroll
    for (int e = 0; e < 8; ++e) { float f = bf2f(v[e]); s += f * f; }
  }
  out[i] = s;
}

// ---------------- q^T k GEMM (WMMA) + L2-dist logits + log-softmax --------------------------
// QT: [B][2048][256] bf16, KT: [B][528][256] bf16 (520 valid rows). One block per (ttile,b).
__global__ void __launch_bounds__(256) attn_softmax_wmma(
    const bf16* __restrict__ QT, const bf16* __restrict__ KT,
    const float* __restrict__ q2, const float* __restrict__ k2,
    float* __restrict__ attn, float* __restrict__ logp)
{
  __shared__ float Lsh[16 * 529];
  __shared__ float red[256];
  const int tid  = threadIdx.x;
  const int lane = tid & 31;
  const int wv   = tid >> 5;
  const int g  = lane >> 4;
  const int mn = lane & 15;
  const int ttile = blockIdx.x;
  const int b     = blockIdx.y;
  const float* q2b = q2 + (size_t)b * 2048 + ttile * 16;
  const float* k2b = k2 + (size_t)b * 520;

  // cache all A fragments (q rows) once per wave: row m = mn, K = 256
  const bf16* Qrow = QT + ((size_t)b * 2048 + ttile * 16 + mn) * 256;
  v16bf af[8];
#pragma unroll
  for (int ks = 0; ks < 8; ++ks) {
    const int a0 = ks * 32 + g * 16;
    af[ks] = cat16(*(const v8bf*)(Qrow + a0), *(const v8bf*)(Qrow + a0 + 8));
  }

  for (int st = wv; st < 33; st += 8) {         // 33 s-tiles of 16 (520 valid)
    const bf16* Krow = KT + ((size_t)b * 528 + st * 16 + mn) * 256;
    if (st + 8 < 33)
      __builtin_prefetch(KT + ((size_t)b * 528 + (st + 8) * 16 + mn) * 256, 0, 0);
    // stage ALL B fragments first; the sched fence pins the loads above the WMMA chain
    v16bf bfr[8];
#pragma unroll
    for (int ks = 0; ks < 8; ++ks) {
      const int k0 = ks * 32 + g * 16;
      bfr[ks] = cat16(*(const v8bf*)(Krow + k0), *(const v8bf*)(Krow + k0 + 8));
    }
    SCHED_FENCE();   // forbid sinking loads into the WMMA chain -> staggered loadcnt waits
    v8f acc = {};
#pragma unroll
    for (int ks = 0; ks < 8; ++ks)
      acc = __builtin_amdgcn_wmma_f32_16x16x32_bf16(false, af[ks], false, bfr[ks],
                                                    (short)0, acc, false, false);
    const int s = st * 16 + mn;
    if (s < 520) {
      const float k2v = k2b[s];
#pragma unroll
      for (int r = 0; r < 8; ++r) {
        const int row = g * 8 + r;
        Lsh[row * 529 + s] = -TEMPERATURE * (q2b[row] + k2v - 2.0f * acc[r]);
      }
    }
  }
  __syncthreads();

  const int row = tid >> 4;
  const int j   = tid & 15;
  float mx = -1e30f;
  for (int s = j; s < 520; s += 16) mx = fmaxf(mx, Lsh[row * 529 + s]);
  red[tid] = mx; __syncthreads();
  for (int off = 8; off > 0; off >>= 1) {
    if (j < off) red[tid] = fmaxf(red[tid], red[tid + off]);
    __syncthreads();
  }
  mx = red[row * 16];
  __syncthreads();
  float sm = 0.0f;
  for (int s = j; s < 520; s += 16) sm += expf(Lsh[row * 529 + s] - mx);
  red[tid] = sm; __syncthreads();
  for (int off = 8; off > 0; off >>= 1) {
    if (j < off) red[tid] += red[tid + off];
    __syncthreads();
  }
  const float lse = mx + logf(red[row * 16]);
  const size_t base = ((size_t)b * 2048 + ttile * 16 + row) * 520;
  for (int s = j; s < 520; s += 16) {
    const float lp = Lsh[row * 529 + s] - lse;
    logp[base + s] = lp;
    attn[base + s] = expf(lp);
  }
}

// ---------------- host orchestration ---------------------------------------------------------
extern "C" void kernel_launch(void* const* d_in, const int* in_sizes, int n_in,
                              void* d_out, int out_size, void* d_ws, size_t ws_size,
                              hipStream_t stream)
{
  const float* queries = (const float*)d_in[0];
  const float* keys    = (const float*)d_in[1];
  const float* kw1 = (const float*)d_in[2];
  const float* kb1 = (const float*)d_in[3];
  const float* kw2 = (const float*)d_in[4];
  const float* kb2 = (const float*)d_in[5];
  const float* qw1 = (const float*)d_in[6];
  const float* qb1 = (const float*)d_in[7];
  const float* qw2 = (const float*)d_in[8];
  const float* qb2 = (const float*)d_in[9];
  const float* qw3 = (const float*)d_in[10];
  const float* qb3 = (const float*)d_in[11];

  const size_t nKeysT = (size_t)32 * 512 * 512;   // keysT [B][512][512]
  const size_t nQryT  = (size_t)32 * 2048 * 80;   // qryT  [B][2048][80]
  const size_t nKW1   = (size_t)256 * 512 * 3;
  const size_t nKW2   = (size_t)256 * 256 * 3;
  const size_t nQW1   = (size_t)256 * 80  * 7;
  const size_t nQW2   = (size_t)256 * 256 * 7;
  const size_t nKB    = (size_t)32 * 528 * 256;   // key-path T-major, 528 padded rows
  const size_t nQB    = (size_t)32 * 2048 * 256;

  bf16* p = (bf16*)d_ws;
  bf16* keysT  = p;  p += nKeysT;
  bf16* qryT   = p;  p += nQryT;
  bf16* kw1P   = p;  p += nKW1;
  bf16* kw2P   = p;  p += nKW2;
  bf16* qw1P   = p;  p += nQW1;
  bf16* qw2P   = p;  p += nQW2;
  bf16* qw3P   = p;  p += nQW2;
  bf16* kbuf1T = p;  p += nKB;
  bf16* kbuf2T = p;  p += nKB;
  bf16* qbufA  = p;  p += nQB;
  bf16* qbufB  = p;  p += nQB;
  float* q2 = (float*)(((uintptr_t)p + 255) & ~(uintptr_t)255);
  float* k2 = q2 + (size_t)32 * 2048;

  auto cdiv = [](long a, long b) { return (int)((a + b - 1) / b); };

  // transpose + convert activations; permute + convert weights
  transpose_f2bf<<<dim3(64, 3, 32),  dim3(32, 8), 0, stream>>>(queries, qryT,  80,  2048);
  transpose_f2bf<<<dim3(16, 16, 32), dim3(32, 8), 0, stream>>>(keys,    keysT, 512, 512);
  permw_f2bf<<<cdiv((long)nKW1, 256), 256, 0, stream>>>(kw1, kw1P, 512, 3, (int)nKW1);
  permw_f2bf<<<cdiv((long)nKW2, 256), 256, 0, stream>>>(kw2, kw2P, 256, 3, (int)nKW2);
  permw_f2bf<<<cdiv((long)nQW1, 256), 256, 0, stream>>>(qw1, qw1P, 80,  7, (int)nQW1);
  permw_f2bf<<<cdiv((long)nQW2, 256), 256, 0, stream>>>(qw2, qw2P, 256, 7, (int)nQW2);
  permw_f2bf<<<cdiv((long)nQW2, 256), 256, 0, stream>>>(qw3, qw3P, 256, 7, (int)nQW2);

  // key path: 512 -> 516 -> 520 valid rows (33 tiles of 16, rows padded to 528)
  conv_lrelu_wmma<3><<<dim3(5, 16, 32),  128, 0, stream>>>(keysT,  kw1P, kb1, kbuf1T, 512, 512, 512, 528, 33);
  conv_lrelu_wmma<3><<<dim3(5, 16, 32),  128, 0, stream>>>(kbuf1T, kw2P, kb2, kbuf2T, 256, 516, 528, 528, 33);
  // query path: 2048 preserved (128 tiles -> 64 tile-pairs)
  conv_lrelu_wmma<7><<<dim3(16, 16, 32), 128, 0, stream>>>(qryT,  qw1P, qb1, qbufA, 80,  2048, 2048, 2048, 128);
  conv_lrelu_wmma<7><<<dim3(16, 16, 32), 128, 0, stream>>>(qbufA, qw2P, qb2, qbufB, 256, 2048, 2048, 2048, 128);
  conv_lrelu_wmma<7><<<dim3(16, 16, 32), 128, 0, stream>>>(qbufB, qw3P, qb3, qbufA, 256, 2048, 2048, 2048, 128);

  sumsqT_kernel<<<cdiv(32L * 2048, 256), 256, 0, stream>>>(qbufA,  q2, 2048, 2048, 32 * 2048);
  sumsqT_kernel<<<cdiv(32L * 520,  256), 256, 0, stream>>>(kbuf2T, k2, 528,  520,  32 * 520);

  float* attnp = (float*)d_out;
  float* lgp   = attnp + (size_t)32 * 2048 * 520;
  attn_softmax_wmma<<<dim3(128, 32), 256, 0, stream>>>(qbufA, kbuf2T, q2, k2, attnp, lgp);
}
